// Glstm_65644280152381
// MI455X (gfx1250) — matrix operand order repaired and checked by reading, not compiled, not run on hardware
//
#include <hip/hip_runtime.h>

#define B_ 16
#define T_ 96
#define N_ 325
#define D_ 64
#define H_ 64
#define G4 256              // 4*H
#define G2 128              // 2*H
#define TSTRIDE (N_ * D_)   // 20800 floats: time stride inside one batch slice

typedef __attribute__((ext_vector_type(16))) _Float16 v16h;
typedef __attribute__((ext_vector_type(8)))  _Float16 v8h;
typedef __attribute__((ext_vector_type(2)))  _Float16 v2h;
typedef __attribute__((ext_vector_type(8)))  float    v8f;

// ---- LDS layout (dynamic) -------------------------------------------------
// [0      ,  32768) UxT   f16[256][64]  (U_x transposed, col-major for B-frags)
// [32768  ,  65536) VxT   f16[256][64]
// [65536  ,  81920) UgT   f16[128][64]
// [81920  ,  98304) VgT   f16[128][64]
// [98304  ,  99328) bx    f32[256]
// [99328  ,  99840) bg    f32[128]
// [99840  , 101888) hbuf  f16[16][64]   (h_{t-1}, row-major)
// [101888 , 118272) gateS f32[16][256]  (sigmoid(gates))
// [118272 , 126464) govS  f32[16][128]  (sigmoid(gov))
#define SMEM_BYTES 126464

// Fast activations: v_exp_f32 + v_rcp_f32 (trans ops co-execute with WMMA pipe).
__device__ __forceinline__ float fsig(float x) {
  return __builtin_amdgcn_rcpf(1.0f + __expf(-x));
}
__device__ __forceinline__ float ftanh(float x) {
  return 1.0f - 2.0f * __builtin_amdgcn_rcpf(1.0f + __expf(2.0f * x));
}

// Build a 16x32 f16 A-fragment straight from a row-major f32 global row.
// p already points at rowBase + ((lane>>4)<<3); kc selects K 0..31 / 32..63.
// Lanes 0-15 get K = kc*32 + {0..7, 16..23}; lanes 16-31 get K = kc*32 + {8..15, 24..31}.
__device__ __forceinline__ v16h loadAfragGlobal(const float* __restrict__ p, int kc) {
  const float* q = p + (kc << 5);
  const float4 a = *(const float4*)(q +  0);
  const float4 b = *(const float4*)(q +  4);
  const float4 c = *(const float4*)(q + 16);
  const float4 d = *(const float4*)(q + 20);
  union { v16h v; v2h h[8]; } u;
  u.h[0] = __builtin_bit_cast(v2h, __builtin_amdgcn_cvt_pkrtz(a.x, a.y));
  u.h[1] = __builtin_bit_cast(v2h, __builtin_amdgcn_cvt_pkrtz(a.z, a.w));
  u.h[2] = __builtin_bit_cast(v2h, __builtin_amdgcn_cvt_pkrtz(b.x, b.y));
  u.h[3] = __builtin_bit_cast(v2h, __builtin_amdgcn_cvt_pkrtz(b.z, b.w));
  u.h[4] = __builtin_bit_cast(v2h, __builtin_amdgcn_cvt_pkrtz(c.x, c.y));
  u.h[5] = __builtin_bit_cast(v2h, __builtin_amdgcn_cvt_pkrtz(c.z, c.w));
  u.h[6] = __builtin_bit_cast(v2h, __builtin_amdgcn_cvt_pkrtz(d.x, d.y));
  u.h[7] = __builtin_bit_cast(v2h, __builtin_amdgcn_cvt_pkrtz(d.z, d.w));
  return u.v;
}

// A-fragment for h (16x64 f16 row-major in LDS).
__device__ __forceinline__ v16h loadHfrag(const _Float16* __restrict__ hb, int lane, int kc) {
  const int row = lane & 15;
  const int kb  = (kc << 5) + ((lane >> 4) << 3);
  union { v16h v; v8h h[2]; } u;
  u.h[0] = *(const v8h*)(hb + row * 64 + kb);
  u.h[1] = *(const v8h*)(hb + row * 64 + kb + 16);
  return u.v;
}

// B-fragment (32x16): weight stored transposed [n][k], K = {8h..8h+7, 16+8h..16+8h+7} per lane.
__device__ __forceinline__ v16h loadBfrag(const _Float16* __restrict__ Wt, int lane, int ntile, int kc) {
  const int col = (ntile << 4) + (lane & 15);
  const int kb  = (kc << 5) + ((lane >> 4) << 3);
  union { v16h v; v8h h[2]; } u;
  u.h[0] = *(const v8h*)(Wt + col * 64 + kb);
  u.h[1] = *(const v8h*)(Wt + col * 64 + kb + 16);
  return u.v;
}

__global__ void __launch_bounds__(128)
glstm_kernel(const float* __restrict__ x,  const float* __restrict__ o_s,
             const float* __restrict__ Ux, const float* __restrict__ Vx,
             const float* __restrict__ bx, const float* __restrict__ Ug,
             const float* __restrict__ Vg, const float* __restrict__ bg,
             float* __restrict__ outHidden, float* __restrict__ outHt,
             float* __restrict__ outCt) {
  extern __shared__ __attribute__((aligned(64))) char smem[];
  _Float16* UxT  = (_Float16*)(smem);
  _Float16* VxT  = (_Float16*)(smem + 32768);
  _Float16* UgT  = (_Float16*)(smem + 65536);
  _Float16* VgT  = (_Float16*)(smem + 81920);
  float*    bxs  = (float*)   (smem + 98304);
  float*    bgs  = (float*)   (smem + 99328);
  _Float16* hbuf = (_Float16*)(smem + 99840);
  float*    gateS = (float*)  (smem + 101888);
  float*    govS  = (float*)  (smem + 118272);

  const int tid  = threadIdx.x;
  const int lane = tid & 31;
  const int w    = tid >> 5;      // wave id 0..3 (wave32)
  const int tile = blockIdx.x;    // 0..324, 16 rows each

  // ---- one-time: transpose + f16-convert weights into LDS ----
  for (int idx = tid; idx < D_ * G4; idx += 128) {
    const int k = idx >> 8, n = idx & (G4 - 1);
    UxT[n * 64 + k] = (_Float16)Ux[idx];
    VxT[n * 64 + k] = (_Float16)Vx[idx];
  }
  for (int idx = tid; idx < D_ * G2; idx += 128) {
    const int k = idx >> 7, n = idx & (G2 - 1);
    UgT[n * 64 + k] = (_Float16)Ug[idx];
    VgT[n * 64 + k] = (_Float16)Vg[idx];
  }
  for (int idx = tid; idx < G4; idx += 128) bxs[idx] = bx[idx];
  for (int idx = tid; idx < G2; idx += 128) bgs[idx] = bg[idx];
  { v8h z = {}; ((v8h*)hbuf)[tid] = z; }   // h0 = 0 (128 * 8 halves = 16x64)
  __syncthreads();

  // ---- per-lane addressing ----
  // A-load rows: R = tile*16 + (lane&15); x index = ((b*T + t)*N + n)*D
  const int rA = tile * 16 + (lane & 15);
  const int bAi = rA / N_;
  const int nAi = rA - bAi * N_;
  const int aBase = bAi * T_ * TSTRIDE + nAi * D_ + ((lane >> 4) << 3);
  const float* xRow = x   + aBase;
  const float* oRow = o_s + aBase;

  // elementwise: wave w owns h-cols [16w, 16w+16); C-layout rows r + 8*(lane>>4)
  const int colE = (w << 4) + (lane & 15);
  int hsOff[8];
#pragma unroll
  for (int r = 0; r < 8; ++r) {
    const int row = r + ((lane >> 4) << 3);
    const int R  = tile * 16 + row;
    const int bb = R / N_;
    const int nn = R - bb * N_;
    hsOff[r] = bb * T_ * TSTRIDE + nn * H_ + colE;  // + t*TSTRIDE at use
  }

  float biasG[4], biasV[2];
#pragma unroll
  for (int nt = 0; nt < 4; ++nt) biasG[nt] = bxs[(((w << 2) + nt) << 4) + (lane & 15)];
#pragma unroll
  for (int nt = 0; nt < 2; ++nt) biasV[nt] = bgs[(((w << 1) + nt) << 4) + (lane & 15)];

  float ct[8];
#pragma unroll
  for (int r = 0; r < 8; ++r) ct[r] = 0.0f;

  // Serial recurrence fenced by barriers each step: unrolling only doubles
  // VGPR pressure (was spilling into VGPR256+ / s_set_vgpr_msb), so keep it rolled.
#pragma unroll 1
  for (int t = 0; t < T_; ++t) {
    const float* xp = xRow + t * TSTRIDE;
    const float* op = oRow + t * TSTRIDE;
    if (t + 1 < T_) {  // warm L2/WGP$ for the next step's streamed rows
      __builtin_prefetch(xp + TSTRIDE, 0, 1);
      __builtin_prefetch(op + TSTRIDE, 0, 1);
    }

    const v16h xA0 = loadAfragGlobal(xp, 0);
    const v16h xA1 = loadAfragGlobal(xp, 1);
    const v16h oA0 = loadAfragGlobal(op, 0);
    const v16h oA1 = loadAfragGlobal(op, 1);
    const v16h hA0 = loadHfrag(hbuf, lane, 0);
    const v16h hA1 = loadHfrag(hbuf, lane, 1);

    // gates = sigmoid(x@Ux + h@Vx + bx): wave w -> N-tiles 4w..4w+3
#pragma unroll
    for (int nt = 0; nt < 4; ++nt) {
      const int gt = (w << 2) + nt;
      const v16h b0 = loadBfrag(UxT, lane, gt, 0);
      const v16h b1 = loadBfrag(UxT, lane, gt, 1);
      const v16h b2 = loadBfrag(VxT, lane, gt, 0);
      const v16h b3 = loadBfrag(VxT, lane, gt, 1);
      v8f c = biasG[nt];
      c = __builtin_amdgcn_wmma_f32_16x16x32_f16(false, xA0, false, b0, (short)0, c, false, false);
      c = __builtin_amdgcn_wmma_f32_16x16x32_f16(false, xA1, false, b1, (short)0, c, false, false);
      c = __builtin_amdgcn_wmma_f32_16x16x32_f16(false, hA0, false, b2, (short)0, c, false, false);
      c = __builtin_amdgcn_wmma_f32_16x16x32_f16(false, hA1, false, b3, (short)0, c, false, false);
#pragma unroll
      for (int r = 0; r < 8; ++r) {
        const int row = r + ((lane >> 4) << 3);
        gateS[row * G4 + (gt << 4) + (lane & 15)] = fsig(c[r]);
      }
    }

    // gov = sigmoid(o_s@Ug + h@Vg + bg): wave w -> N-tiles 2w, 2w+1
#pragma unroll
    for (int nt = 0; nt < 2; ++nt) {
      const int gt = (w << 1) + nt;
      const v16h b0 = loadBfrag(UgT, lane, gt, 0);
      const v16h b1 = loadBfrag(UgT, lane, gt, 1);
      const v16h b2 = loadBfrag(VgT, lane, gt, 0);
      const v16h b3 = loadBfrag(VgT, lane, gt, 1);
      v8f c = biasV[nt];
      c = __builtin_amdgcn_wmma_f32_16x16x32_f16(false, oA0, false, b0, (short)0, c, false, false);
      c = __builtin_amdgcn_wmma_f32_16x16x32_f16(false, oA1, false, b1, (short)0, c, false, false);
      c = __builtin_amdgcn_wmma_f32_16x16x32_f16(false, hA0, false, b2, (short)0, c, false, false);
      c = __builtin_amdgcn_wmma_f32_16x16x32_f16(false, hA1, false, b3, (short)0, c, false, false);
#pragma unroll
      for (int r = 0; r < 8; ++r) {
        const int row = r + ((lane >> 4) << 3);
        govS[row * G2 + (gt << 4) + (lane & 15)] = fsig(c[r]);
      }
    }

    __syncthreads();  // gates/gov visible; h-buffer free to overwrite

    // cell update on owned 16x16 column slice (ct resident in VGPRs)
#pragma unroll
    for (int r = 0; r < 8; ++r) {
      const int row = r + ((lane >> 4) << 3);
      const float ft = gateS[row * G4 +       colE];
      const float it = gateS[row * G4 +  64 + colE];
      const float ot = gateS[row * G4 + 128 + colE];
      const float ut = gateS[row * G4 + 192 + colE];
      const float gf = govS[row * G2 +       colE];
      const float gu = govS[row * G2 +  64 + colE];
      const float cc = gf * ft * ct[r] + gu * it * ut;
      ct[r] = cc;
      const float hh = ot * ftanh(cc);
      // write-once stream: non-temporal so it doesn't evict weights / future x rows
      __builtin_nontemporal_store(hh, &outHidden[hsOff[r] + t * TSTRIDE]);
      hbuf[row * 64 + colE] = (_Float16)hh;
      if (t == T_ - 1) {
        const int R = tile * 16 + row;
        outHt[R * 64 + colE] = hh;
        outCt[R * 64 + colE] = cc;
      }
    }
    __syncthreads();  // new h visible before next step's A-frag reads
  }
}

extern "C" void kernel_launch(void* const* d_in, const int* in_sizes, int n_in,
                              void* d_out, int out_size, void* d_ws, size_t ws_size,
                              hipStream_t stream) {
  (void)in_sizes; (void)n_in; (void)out_size; (void)d_ws; (void)ws_size;
  const float* x  = (const float*)d_in[0];
  const float* os = (const float*)d_in[1];
  const float* Ux = (const float*)d_in[2];
  const float* Vx = (const float*)d_in[3];
  const float* bx = (const float*)d_in[4];
  const float* Ug = (const float*)d_in[5];
  const float* Vg = (const float*)d_in[6];
  const float* bg = (const float*)d_in[7];

  float* out       = (float*)d_out;
  float* outHidden = out;                                      // (B,T,N,H)
  float* outHt     = out + (size_t)B_ * T_ * N_ * H_;          // (B,N,H)
  float* outCt     = outHt + (size_t)B_ * N_ * H_;             // (B,N,H)

  glstm_kernel<<<dim3(N_), dim3(128), SMEM_BYTES, stream>>>(
      x, os, Ux, Vx, bx, Ug, Vg, bg, outHidden, outHt, outCt);
}
